// CPM3bmtCrossAttention_1022202216943
// MI455X (gfx1250) — matrix operand order repaired and checked by reading, not compiled, not run on hardware
//
#include <hip/hip_runtime.h>
#include <cstdint>
#include <cstddef>

// ---------- constants (match reference) ----------
#define B_   2
#define LQ_  1024
#define LK_  1024
#define D_   4096
#define H_   32
#define DH_  128
#define HD_  (H_ * DH_)   // 4096

typedef __attribute__((ext_vector_type(16))) __bf16        bf16x16;
typedef __attribute__((ext_vector_type(8)))  float         f32x8;
typedef __attribute__((ext_vector_type(4)))  unsigned int  u32x4;
typedef int i32x4v __attribute__((vector_size(16)));       // matches builtin param pointee

#define AS1 __attribute__((address_space(1)))
#define AS3 __attribute__((address_space(3)))

// CDNA5 async global->LDS copies (ASYNCcnt) when the toolchain exposes them.
#if defined(__has_builtin)
#  if __has_builtin(__builtin_amdgcn_global_load_async_to_lds_b128)
#    define USE_ASYNC_LDS 1
#  endif
#endif
#ifndef USE_ASYNC_LDS
#  define USE_ASYNC_LDS 0
#endif

__device__ __forceinline__ void wait_asynccnt0() {
#if USE_ASYNC_LDS
#  if __has_builtin(__builtin_amdgcn_s_wait_asynccnt)
  __builtin_amdgcn_s_wait_asynccnt(0);
#  else
  asm volatile("s_wait_asynccnt 0" ::: "memory");
#  endif
#endif
}

// ---------- helpers ----------
__device__ __forceinline__ unsigned short f32_to_bf16(float f) {
  union { float f; uint32_t u; } v; v.f = f;
  uint32_t u = v.u;
  uint32_t r = u + 0x7FFFu + ((u >> 16) & 1u);   // round-to-nearest-even
  return (unsigned short)(r >> 16);
}

// A-matrix fragment (16x32 bf16), ISA layout:
// lane<16 : row=lane,  K 0..7 and 16..23 ; lane>=16: row=lane-16, K 8..15 and 24..31
__device__ __forceinline__ bf16x16 load_frag_a(const unsigned short* base, int ld) {
  const int lane = threadIdx.x & 31;
  const int row  = lane & 15;
  const int hi   = lane >> 4;
  const unsigned short* p = base + row * ld + hi * 8;
  union { bf16x16 v; u32x4 q[2]; } u;
  u.q[0] = *reinterpret_cast<const u32x4*>(p);
  u.q[1] = *reinterpret_cast<const u32x4*>(p + 16);
  return u.v;
}

// B-matrix fragment (32x16 bf16), element (k,n) = base[n*ld + k]:
// lane<16 : col=lane, K 0..7 and 16..23 ; lane>=16 : col=lane-16, K 8..15 and 24..31
__device__ __forceinline__ bf16x16 load_frag_b(const unsigned short* base, int ld) {
  const int lane = threadIdx.x & 31;
  const int n    = lane & 15;
  const int hi   = lane >> 4;
  const unsigned short* p = base + n * ld + hi * 8;
  union { bf16x16 v; u32x4 q[2]; } u;
  u.q[0] = *reinterpret_cast<const u32x4*>(p);
  u.q[1] = *reinterpret_cast<const u32x4*>(p + 16);
  return u.v;
}

// ---------- elementwise fp32 -> bf16 ----------
__global__ __launch_bounds__(256)
void cvt_f32_bf16_kernel(const float* __restrict__ in, unsigned short* __restrict__ out, int n) {
  int i = blockIdx.x * 256 + threadIdx.x;
  if (i < n) out[i] = f32_to_bf16(in[i]);
}

// ---------- RMSNorm(hidden) -> bf16 ----------
__global__ __launch_bounds__(256)
void rmsnorm_bf16_kernel(const float* __restrict__ x, const float* __restrict__ w,
                         unsigned short* __restrict__ out) {
  __shared__ float red[256];
  const int row = blockIdx.x;
  const float* xr = x + (size_t)row * D_;
  float ss = 0.f;
  for (int i = threadIdx.x; i < D_; i += 256) { float v = xr[i]; ss += v * v; }
  red[threadIdx.x] = ss;
  __syncthreads();
  for (int s = 128; s > 0; s >>= 1) {
    if (threadIdx.x < s) red[threadIdx.x] += red[threadIdx.x + s];
    __syncthreads();
  }
  const float inv = rsqrtf(red[0] / (float)D_ + 1e-5f);
  for (int i = threadIdx.x; i < D_; i += 256)
    out[(size_t)row * D_ + i] = f32_to_bf16(xr[i] * inv * w[i]);
}

// ---------- GEMM: C[M,N] = A[M,K] * W[N,K]^T (bf16), 128x128x32 tiles ----------
// Double-buffered LDS pipeline, one barrier per K-step.
// FP32OUT: fp32 out + residual ; else bf16 out.
template <bool FP32OUT>
__global__ __launch_bounds__(256)
void gemm_bf16_wmma(const unsigned short* __restrict__ A,
                    const unsigned short* __restrict__ W,
                    unsigned short* __restrict__ Cb,
                    float* __restrict__ Cf,
                    const float* __restrict__ resid,
                    int M, int N, int K) {
  __shared__ unsigned short As[2][128 * 40];   // row stride 40 elems: bank-conflict-free, 16B aligned
  __shared__ unsigned short Ws[2][128 * 40];

  const int m0 = blockIdx.y * 128;
  const int n0 = blockIdx.x * 128;
  const int t    = threadIdx.x;
  const int wave = t >> 5;
  const int wm   = wave >> 2;        // 0..1 -> 64 rows
  const int wn   = wave & 3;         // 0..3 -> 32 cols
  const int lane = t & 31;
  const int hi   = lane >> 4;
  const int colb = lane & 15;

  const int srow = t >> 1;           // staging: 2 threads per row, 16 elems each
  const int scol = (t & 1) * 16;

  const unsigned short* gA = A + (size_t)(m0 + srow) * K + scol;
  const unsigned short* gW = W + (size_t)(n0 + srow) * K + scol;

#if USE_ASYNC_LDS
  auto issue = [&](int k0, int buf) {
    __builtin_amdgcn_global_load_async_to_lds_b128(
        (AS1 i32x4v*)(gA + k0), (AS3 i32x4v*)&As[buf][srow * 40 + scol], 0, 0);
    __builtin_amdgcn_global_load_async_to_lds_b128(
        (AS1 i32x4v*)(gA + k0 + 8), (AS3 i32x4v*)&As[buf][srow * 40 + scol + 8], 0, 0);
    __builtin_amdgcn_global_load_async_to_lds_b128(
        (AS1 i32x4v*)(gW + k0), (AS3 i32x4v*)&Ws[buf][srow * 40 + scol], 0, 0);
    __builtin_amdgcn_global_load_async_to_lds_b128(
        (AS1 i32x4v*)(gW + k0 + 8), (AS3 i32x4v*)&Ws[buf][srow * 40 + scol + 8], 0, 0);
  };
  issue(0, 0);
#else
  u32x4 pa0, pa1, pw0, pw1;
  auto fetch = [&](int k0) {
    const u32x4* a = reinterpret_cast<const u32x4*>(gA + k0);
    const u32x4* w = reinterpret_cast<const u32x4*>(gW + k0);
    pa0 = a[0]; pa1 = a[1]; pw0 = w[0]; pw1 = w[1];
  };
  fetch(0);
#endif

  f32x8 acc[4][2] = {};
  const int nsteps = K / 32;

  for (int i = 0; i < nsteps; ++i) {
    const int buf = i & 1;
    // ---- commit tile i into LDS[buf] ----
#if USE_ASYNC_LDS
    wait_asynccnt0();
#else
    {
      u32x4* da = reinterpret_cast<u32x4*>(&As[buf][srow * 40 + scol]);
      u32x4* dw = reinterpret_cast<u32x4*>(&Ws[buf][srow * 40 + scol]);
      da[0] = pa0; da[1] = pa1;
      dw[0] = pw0; dw[1] = pw1;
    }
#endif
    __syncthreads();

    // ---- start fetching tile i+1 (hidden under the WMMAs below) ----
#if USE_ASYNC_LDS
    if (i + 1 < nsteps) issue((i + 1) * 32, buf ^ 1);
#else
    if (i + 1 < nsteps) fetch((i + 1) * 32);
#endif

    // ---- compute from LDS[buf] ----
    bf16x16 af[4];
#pragma unroll
    for (int fm = 0; fm < 4; ++fm) af[fm] = load_frag_a(&As[buf][(wm * 64 + fm * 16) * 40], 40);
    bf16x16 bw[2];
#pragma unroll
    for (int fn = 0; fn < 2; ++fn) bw[fn] = load_frag_b(&Ws[buf][(wn * 32 + fn * 16) * 40], 40);

#pragma unroll
    for (int fm = 0; fm < 4; ++fm)
#pragma unroll
      for (int fn = 0; fn < 2; ++fn)
        acc[fm][fn] = __builtin_amdgcn_wmma_f32_16x16x32_bf16(
            false, af[fm], false, bw[fn], (short)0, acc[fm][fn], false, false);
  }

#pragma unroll
  for (int fm = 0; fm < 4; ++fm)
#pragma unroll
    for (int fn = 0; fn < 2; ++fn)
#pragma unroll
      for (int j = 0; j < 8; ++j) {
        const int r = m0 + wm * 64 + fm * 16 + j + hi * 8;
        const int c = n0 + wn * 32 + fn * 16 + colb;
        const size_t idx = (size_t)r * N + c;
        const float v = acc[fm][fn][j];
        if (FP32OUT) Cf[idx] = resid[idx] + v;
        else         Cb[idx] = f32_to_bf16(v);
      }
}

// ---------- flash attention: 4 waves/block, wave = 16 q-rows, 32-key steps ----------
// Double-buffered K/V tiles, one barrier per step.
__global__ __launch_bounds__(128)
void flash_attn_wmma(const unsigned short* __restrict__ Q,
                     const unsigned short* __restrict__ Kc,
                     const unsigned short* __restrict__ Vc,
                     const float* __restrict__ bias,           // [H, LQ, LK] fp32
                     const unsigned char* __restrict__ mask,   // [B, LQ, LK] bool
                     unsigned short* __restrict__ Out) {       // [B*LQ, H*DH] bf16
  __shared__ unsigned short Kt[2][32 * 136];    // K tile [key][dh], stride 136 (272B, 16B-mult)
  __shared__ unsigned short Vt[2][128 * 40];    // V tile transposed [dh][key], stride 40
  __shared__ unsigned short Pw[4 * 16 * 40];    // per-wave P scratch [16][40]

  const int bi   = blockIdx.z;
  const int h    = blockIdx.y;
  const int wave = threadIdx.x >> 5;
  const int lane = threadIdx.x & 31;
  const int hi   = lane >> 4;
  const int colb = lane & 15;
  const int q0   = blockIdx.x * 64 + wave * 16;

  // Q fragments for this wave's 16 rows, direct from global (row stride HD_)
  bf16x16 qa[4];
#pragma unroll
  for (int c = 0; c < 4; ++c)
    qa[c] = load_frag_a(Q + (size_t)(bi * LQ_ + q0) * HD_ + h * DH_ + c * 32, HD_);

  f32x8 Oacc[8] = {};
  float mrow[8], lrow[8];
#pragma unroll
  for (int j = 0; j < 8; ++j) { mrow[j] = -__builtin_inff(); lrow[j] = 0.f; }

  const int skey = threadIdx.x >> 2;         // staging: 0..31 keys
  const int sdh  = (threadIdx.x & 3) * 32;   // 4 threads per key row

  const unsigned short* gK = Kc + (size_t)(bi * LK_ + skey) * HD_ + h * DH_ + sdh;
  const unsigned short* gV = Vc + (size_t)(bi * LK_ + skey) * HD_ + h * DH_ + sdh;

  union VU { u32x4 q[4]; unsigned short s[32]; } vreg;
  auto fetchV = [&](int k0) {
    const u32x4* g = reinterpret_cast<const u32x4*>(gV + (size_t)k0 * HD_);
    vreg.q[0] = g[0]; vreg.q[1] = g[1]; vreg.q[2] = g[2]; vreg.q[3] = g[3];
  };

#if USE_ASYNC_LDS
  auto issueK = [&](int k0, int buf) {
#pragma unroll
    for (int jj = 0; jj < 4; ++jj)
      __builtin_amdgcn_global_load_async_to_lds_b128(
          (AS1 i32x4v*)(gK + (size_t)k0 * HD_ + jj * 8),
          (AS3 i32x4v*)&Kt[buf][skey * 136 + sdh + jj * 8], 0, 0);
  };
  issueK(0, 0);
#else
  u32x4 kreg[4];
  auto fetchK = [&](int k0) {
    const u32x4* g = reinterpret_cast<const u32x4*>(gK + (size_t)k0 * HD_);
    kreg[0] = g[0]; kreg[1] = g[1]; kreg[2] = g[2]; kreg[3] = g[3];
  };
  fetchK(0);
#endif
  fetchV(0);

  for (int i = 0; i < LK_ / 32; ++i) {
    const int k0  = i * 32;
    const int buf = i & 1;

    // ---- commit tile i into LDS[buf] ----
#if USE_ASYNC_LDS
    wait_asynccnt0();
#else
    {
      u32x4* dk = reinterpret_cast<u32x4*>(&Kt[buf][skey * 136 + sdh]);
      dk[0] = kreg[0]; dk[1] = kreg[1]; dk[2] = kreg[2]; dk[3] = kreg[3];
    }
#endif
#pragma unroll
    for (int v = 0; v < 32; ++v) Vt[buf][(sdh + v) * 40 + skey] = vreg.s[v];
    __syncthreads();

    // ---- start fetching tile i+1 ----
    if (i + 1 < LK_ / 32) {
#if USE_ASYNC_LDS
      issueK(k0 + 32, buf ^ 1);
#else
      fetchK(k0 + 32);
#endif
      fetchV(k0 + 32);
    }

    // ---- S = Q * K^T (two 16x16 f32 tiles), + bias, mask ----
    float sv[2][8];
#pragma unroll
    for (int f = 0; f < 2; ++f) {
      f32x8 acc = {};
#pragma unroll
      for (int c = 0; c < 4; ++c) {
        bf16x16 kb = load_frag_b(&Kt[buf][(f * 16) * 136 + c * 32], 136);
        acc = __builtin_amdgcn_wmma_f32_16x16x32_bf16(
            false, qa[c], false, kb, (short)0, acc, false, false);
      }
#pragma unroll
      for (int j = 0; j < 8; ++j) {
        const int qr = q0 + j + hi * 8;
        const int kc = k0 + f * 16 + colb;
        float s = acc[j] + bias[(size_t)(h * LQ_ + qr) * LK_ + kc];
        if (!mask[(size_t)(bi * LQ_ + qr) * LK_ + kc]) s = -1e30f;
        sv[f][j] = s;
      }
    }

    // ---- online softmax (row = j + 8*hi; reduce across the 16 lanes of each half) ----
    float alpha[8], mnew[8];
#pragma unroll
    for (int j = 0; j < 8; ++j) {
      float mx = fmaxf(sv[0][j], sv[1][j]);
#pragma unroll
      for (int d = 1; d < 16; d <<= 1) mx = fmaxf(mx, __shfl_xor(mx, d, 32));
      const float mn = fmaxf(mrow[j], mx);
      alpha[j] = __expf(mrow[j] - mn);
      mnew[j]  = mn;
      mrow[j]  = mn;
    }

    unsigned short* pw = &Pw[wave * 16 * 40];
    float rs[8];
#pragma unroll
    for (int j = 0; j < 8; ++j) rs[j] = 0.f;
#pragma unroll
    for (int f = 0; f < 2; ++f)
#pragma unroll
      for (int j = 0; j < 8; ++j) {
        const float p = __expf(sv[f][j] - mnew[j]);   // masked entries underflow to 0
        rs[j] += p;
        pw[(j + hi * 8) * 40 + f * 16 + colb] = f32_to_bf16(p);
      }
#pragma unroll
    for (int j = 0; j < 8; ++j) {
      float r = rs[j];
#pragma unroll
      for (int d = 1; d < 16; d <<= 1) r += __shfl_xor(r, d, 32);
      lrow[j] = lrow[j] * alpha[j] + r;
#pragma unroll
      for (int dc = 0; dc < 8; ++dc) Oacc[dc][j] *= alpha[j];
    }

    // wave-local LDS visibility before re-reading P as an A fragment (split DS counter)
    asm volatile("s_wait_dscnt 0" ::: "memory");

    // ---- O += P * V ----
    bf16x16 pa = load_frag_a(pw, 40);
#pragma unroll
    for (int dc = 0; dc < 8; ++dc) {
      bf16x16 vb = load_frag_b(&Vt[buf][(dc * 16) * 40], 40);
      Oacc[dc] = __builtin_amdgcn_wmma_f32_16x16x32_bf16(
          false, pa, false, vb, (short)0, Oacc[dc], false, false);
    }
  }

  // ---- normalize and store bf16 attn ----
#pragma unroll
  for (int j = 0; j < 8; ++j) {
    const float inv = lrow[j] > 0.f ? 1.f / lrow[j] : 0.f;
    const int qr = q0 + j + hi * 8;
#pragma unroll
    for (int dc = 0; dc < 8; ++dc)
      Out[(size_t)(bi * LQ_ + qr) * HD_ + h * DH_ + dc * 16 + colb] =
          f32_to_bf16(Oacc[dc][j] * inv);
  }
}

// ---------- host launcher ----------
extern "C" void kernel_launch(void* const* d_in, const int* in_sizes, int n_in,
                              void* d_out, int out_size, void* d_ws, size_t ws_size,
                              hipStream_t stream) {
  (void)in_sizes; (void)n_in; (void)out_size; (void)ws_size;

  const float*         hidden = (const float*)d_in[0];
  const float*         kv     = (const float*)d_in[1];
  const unsigned char* mask   = (const unsigned char*)d_in[2];
  const float*         bias   = (const float*)d_in[3];
  const float*         lnw    = (const float*)d_in[4];
  const float*         wq     = (const float*)d_in[5];
  const float*         wk     = (const float*)d_in[6];
  const float*         wv     = (const float*)d_in[7];
  const float*         wo     = (const float*)d_in[8];
  float* out = (float*)d_out;

  char* ws = (char*)d_ws;
  const size_t szAct = (size_t)B_ * LQ_ * D_ * 2;  // 16 MiB bf16 activations
  const size_t szW   = (size_t)D_ * D_ * 2;        // 32 MiB bf16 weights
  unsigned short* xn  = (unsigned short*)(ws);
  unsigned short* kvb = (unsigned short*)(ws + 1 * szAct);
  unsigned short* qb  = (unsigned short*)(ws + 2 * szAct);
  unsigned short* kb  = (unsigned short*)(ws + 3 * szAct);
  unsigned short* vb  = (unsigned short*)(ws + 4 * szAct);
  unsigned short* ab  = (unsigned short*)(ws + 5 * szAct);
  unsigned short* wqb = (unsigned short*)(ws + 6 * szAct);
  unsigned short* wkb = (unsigned short*)(ws + 6 * szAct + 1 * szW);
  unsigned short* wvb = (unsigned short*)(ws + 6 * szAct + 2 * szW);
  unsigned short* wob = (unsigned short*)(ws + 6 * szAct + 3 * szW);

  // 1) precision conversion (weights fit in 192MB L2 as bf16 -> GEMMs stream from L2)
  const int nW = D_ * D_;
  cvt_f32_bf16_kernel<<<dim3((nW + 255) / 256), 256, 0, stream>>>(wq, wqb, nW);
  cvt_f32_bf16_kernel<<<dim3((nW + 255) / 256), 256, 0, stream>>>(wk, wkb, nW);
  cvt_f32_bf16_kernel<<<dim3((nW + 255) / 256), 256, 0, stream>>>(wv, wvb, nW);
  cvt_f32_bf16_kernel<<<dim3((nW + 255) / 256), 256, 0, stream>>>(wo, wob, nW);
  const int nKV = B_ * LK_ * D_;
  cvt_f32_bf16_kernel<<<dim3((nKV + 255) / 256), 256, 0, stream>>>(kv, kvb, nKV);

  // 2) RMSNorm(hidden) -> bf16
  rmsnorm_bf16_kernel<<<dim3(B_ * LQ_), 256, 0, stream>>>(hidden, lnw, xn);

  // 3) projections (WMMA bf16 GEMMs)
  dim3 ggrid(D_ / 128, (B_ * LQ_) / 128);   // (32, 16)
  gemm_bf16_wmma<false><<<ggrid, 256, 0, stream>>>(xn,  wqb, qb, nullptr, nullptr, B_ * LQ_, HD_, D_);
  gemm_bf16_wmma<false><<<ggrid, 256, 0, stream>>>(kvb, wkb, kb, nullptr, nullptr, B_ * LK_, HD_, D_);
  gemm_bf16_wmma<false><<<ggrid, 256, 0, stream>>>(kvb, wvb, vb, nullptr, nullptr, B_ * LK_, HD_, D_);

  // 4) flash attention (WMMA QK^T and PV, online softmax)
  dim3 agrid(LQ_ / 64, H_, B_);
  flash_attn_wmma<<<agrid, 128, 0, stream>>>(qb, kb, vb, bias, mask, ab);

  // 5) output projection + residual -> fp32 d_out
  gemm_bf16_wmma<true><<<ggrid, 256, 0, stream>>>(ab, wob, nullptr, out, hidden, B_ * LQ_, D_, HD_);
}
